// MLP_24498493456624
// MI455X (gfx1250) — compile-verified, matrix operand order checked by reading, not run on hardware
//
#include <hip/hip_runtime.h>
#include <hip/hip_bf16.h>

typedef __attribute__((ext_vector_type(16))) __bf16 v16bf;
typedef __attribute__((ext_vector_type(8)))  __bf16 v8bf;
typedef __attribute__((ext_vector_type(8)))  float  v8f;
typedef __attribute__((ext_vector_type(4)))  int    v4i_t;

#define B_DIM 8192
#define N_DIM 4096
#define R_DIM 16
#define O_DIM 4096

// ---------------------------------------------------------------------------
// gfx1250 async global->LDS copy of 16 bytes per lane (ASYNCcnt-tracked),
// with a compile-guaranteed fallback to plain load + ds_store.
// ---------------------------------------------------------------------------
__device__ __forceinline__ void async_cp16(const __bf16* g, __bf16* l) {
#if __has_builtin(__builtin_amdgcn_global_load_async_to_lds_b128)
    __builtin_amdgcn_global_load_async_to_lds_b128(
        (__attribute__((address_space(1))) v4i_t*)g,
        (__attribute__((address_space(3))) v4i_t*)l,
        0, 0);
#else
    *(uint4*)l = *(const uint4*)g;
#endif
}

__device__ __forceinline__ void wait_async_all() {
#if __has_builtin(__builtin_amdgcn_global_load_async_to_lds_b128)
#if __has_builtin(__builtin_amdgcn_s_wait_asynccnt)
    __builtin_amdgcn_s_wait_asynccnt(0);
#else
    asm volatile("s_wait_asynccnt 0" ::: "memory");
#endif
#endif
}

// ---------------------------------------------------------------------------
// Kernel 1: xH[b][r] = sum_n x[b][n] * H[r][n]   (one wave per row b)
// x is streamed once -> non-temporal loads keep L2 free.
// ---------------------------------------------------------------------------
__global__ __launch_bounds__(256) void xh_kernel(const float* __restrict__ x,
                                                 const float* __restrict__ H,
                                                 float* __restrict__ xH) {
    int wave = threadIdx.x >> 5;
    int lane = threadIdx.x & 31;
    int b = blockIdx.x * 8 + wave;          // gridDim.x = B/8
    const float* xr = x + (size_t)b * N_DIM;
    float acc[R_DIM];
#pragma unroll
    for (int r = 0; r < R_DIM; ++r) acc[r] = 0.0f;

    for (int i = lane; i < N_DIM; i += 32) {
        float xv = __builtin_nontemporal_load(xr + i);
#pragma unroll
        for (int r = 0; r < R_DIM; ++r)
            acc[r] += xv * H[r * N_DIM + i];
    }
#pragma unroll
    for (int r = 0; r < R_DIM; ++r) {
#pragma unroll
        for (int off = 16; off >= 1; off >>= 1)
            acc[r] += __shfl_xor(acc[r], off, 32);
    }
    if (lane == 0) {
#pragma unroll
        for (int r = 0; r < R_DIM; ++r)
            xH[(size_t)b * R_DIM + r] = acc[r];
    }
}

// ---------------------------------------------------------------------------
// Kernel 2: W2T[n][k] = bf16(W2[k][n])  (tiled transpose + convert)
// block (32,8), grid (N/32, K/32)
// ---------------------------------------------------------------------------
__global__ __launch_bounds__(256) void w2t_kernel(const float* __restrict__ W2,
                                                  __bf16* __restrict__ W2T) {
    __shared__ float tile[32][33];
    int tx = threadIdx.x, ty = threadIdx.y;
    int kBase = blockIdx.y * 32, nBase = blockIdx.x * 32;
#pragma unroll
    for (int j = 0; j < 32; j += 8)
        tile[ty + j][tx] = W2[(size_t)(kBase + ty + j) * O_DIM + nBase + tx];
    __syncthreads();
#pragma unroll
    for (int j = 0; j < 32; j += 8)
        W2T[(size_t)(nBase + ty + j) * N_DIM + kBase + tx] = (__bf16)tile[tx][ty + j];
}

// ---------------------------------------------------------------------------
// Kernel 3: h[b][n] = bf16(relu(sum_r xH[b][r]*G[r][n] + b1[n]))
// ---------------------------------------------------------------------------
__global__ __launch_bounds__(256) void h_kernel(const float* __restrict__ xH,
                                                const float* __restrict__ G,
                                                const float* __restrict__ b1,
                                                __bf16* __restrict__ h) {
    size_t idx = (size_t)blockIdx.x * blockDim.x + threadIdx.x;  // over B*N
    int n = (int)(idx & (N_DIM - 1));
    size_t b = idx >> 12;                                        // N==4096
    const float* xh = xH + b * R_DIM;
    float s = b1[n];
#pragma unroll
    for (int r = 0; r < R_DIM; ++r)
        s += xh[r] * G[r * N_DIM + n];
    s = s > 0.0f ? s : 0.0f;
    h[idx] = (__bf16)s;
}

// ---------------------------------------------------------------------------
// Kernel 4: Y[m][n] = sum_k A[m][k] * Bt[n][k] + b2[n]
//   A  = h   bf16 [M=8192][K=4096] row-major
//   Bt = W2T bf16 [N=4096][K=4096] row-major (K-contiguous per output col)
// Block: 256 threads (8 waves), 128x256 tile, BK=64, double-buffered LDS
// filled by async global->LDS copies (ASYNCcnt) overlapped with WMMA compute.
// Wave (wRow, wCol) in 2x4 grid computes 64x64 via 4x4 WMMA accumulators
// (16 ds_load_b128 per 16 WMMAs per k-step).
// ---------------------------------------------------------------------------
__global__ __launch_bounds__(256) void gemm_kernel(const __bf16* __restrict__ A,
                                                   const __bf16* __restrict__ Bt,
                                                   const float* __restrict__ b2,
                                                   float* __restrict__ Y) {
    const int K = N_DIM;   // 4096
    const int N = O_DIM;   // 4096
    const int TILES = K / 64;

    __shared__ __align__(16) __bf16 As[2][128][72];  // 2 x (128 m x 64 k + pad)
    __shared__ __align__(16) __bf16 Bs[2][256][72];  // 2 x (256 n x 64 k + pad)

    int tid   = threadIdx.x;
    int mBase = blockIdx.y * 128;
    int nBase = blockIdx.x * 256;
    int w     = tid >> 5;
    int lane  = tid & 31;
    int wRow  = w & 1;          // 0..1  -> 64-row strip
    int wCol  = w >> 1;         // 0..3  -> 64-col strip
    int lrow  = lane & 15;
    int lhalf = lane >> 4;

    v8f acc[4][4] = {};

    // per-thread addressing for cooperative tile copies:
    // A: 128 rows * 8 chunks = 1024 -> 4/thread ; B: 256 rows * 8 = 2048 -> 8/thread
    int arow[4], acol[4], brow[8], bcol[8];
#pragma unroll
    for (int i = 0; i < 4; ++i) {
        int idx = tid + i * 256;
        arow[i] = idx >> 3;
        acol[i] = (idx & 7) << 3;
    }
#pragma unroll
    for (int i = 0; i < 8; ++i) {
        int idx = tid + i * 256;
        brow[i] = idx >> 3;
        bcol[i] = (idx & 7) << 3;
    }

    // prologue: stage tile 0 into buffer 0
#pragma unroll
    for (int i = 0; i < 4; ++i)
        async_cp16(A + (size_t)(mBase + arow[i]) * K + acol[i], &As[0][arow[i]][acol[i]]);
#pragma unroll
    for (int i = 0; i < 8; ++i)
        async_cp16(Bt + (size_t)(nBase + brow[i]) * K + bcol[i], &Bs[0][brow[i]][bcol[i]]);

    for (int t = 0; t < TILES; ++t) {
        int buf = t & 1;
        wait_async_all();      // own tile-t async copies have landed in LDS
        __syncthreads();       // all waves' copies landed; all done reading buf^1

        if (t + 1 < TILES) {   // stage tile t+1 into the other buffer (overlaps compute)
            int k0 = (t + 1) * 64;
            int nb = buf ^ 1;
#pragma unroll
            for (int i = 0; i < 4; ++i)
                async_cp16(A + (size_t)(mBase + arow[i]) * K + k0 + acol[i],
                           &As[nb][arow[i]][acol[i]]);
#pragma unroll
            for (int i = 0; i < 8; ++i)
                async_cp16(Bt + (size_t)(nBase + brow[i]) * K + k0 + bcol[i],
                           &Bs[nb][brow[i]][bcol[i]]);
        }

#pragma unroll
        for (int kk = 0; kk < 64; kk += 32) {
            int koff = kk + lhalf * 8;   // ISA 16-bit A layout: lane>>4 selects K half
            v16bf a[4], bb[4];
#pragma unroll
            for (int r = 0; r < 4; ++r) {
                v8bf lo = *(const v8bf*)(&As[buf][wRow * 64 + r * 16 + lrow][koff]);
                v8bf hi = *(const v8bf*)(&As[buf][wRow * 64 + r * 16 + lrow][koff + 16]);
                a[r] = __builtin_shufflevector(lo, hi, 0, 1, 2, 3, 4, 5, 6, 7,
                                               8, 9, 10, 11, 12, 13, 14, 15);
            }
#pragma unroll
            for (int c = 0; c < 4; ++c) {
                v8bf lo = *(const v8bf*)(&Bs[buf][wCol * 64 + c * 16 + lrow][koff]);
                v8bf hi = *(const v8bf*)(&Bs[buf][wCol * 64 + c * 16 + lrow][koff + 16]);
                bb[c] = __builtin_shufflevector(lo, hi, 0, 1, 2, 3, 4, 5, 6, 7,
                                                8, 9, 10, 11, 12, 13, 14, 15);
            }
#pragma unroll
            for (int r = 0; r < 4; ++r)
#pragma unroll
                for (int c = 0; c < 4; ++c)
                    acc[r][c] = __builtin_amdgcn_wmma_f32_16x16x32_bf16(
                        false, a[r], false, bb[c], (short)0, acc[r][c], false, false);
        }
    }

    // epilogue: D layout -> VGPR v: M = v + 8*(lane>>4), N = lane&15
    // Y is write-once, never re-read -> non-temporal stores keep h/W2T in L2.
#pragma unroll
    for (int r = 0; r < 4; ++r) {
#pragma unroll
        for (int c = 0; c < 4; ++c) {
            int gm0 = mBase + wRow * 64 + r * 16 + lhalf * 8;
            int gn  = nBase + wCol * 64 + c * 16 + lrow;
            float bias = b2[gn];
#pragma unroll
            for (int v = 0; v < 8; ++v)
                __builtin_nontemporal_store(acc[r][c][v] + bias,
                                            Y + (size_t)(gm0 + v) * N + gn);
        }
    }
}

// ---------------------------------------------------------------------------
extern "C" void kernel_launch(void* const* d_in, const int* in_sizes, int n_in,
                              void* d_out, int out_size, void* d_ws, size_t ws_size,
                              hipStream_t stream) {
    const float* x  = (const float*)d_in[0];
    const float* G  = (const float*)d_in[1];
    const float* H  = (const float*)d_in[2];
    const float* b1 = (const float*)d_in[3];
    const float* W2 = (const float*)d_in[4];
    const float* b2 = (const float*)d_in[5];
    float* y = (float*)d_out;

    char* ws = (char*)d_ws;
    __bf16* hbuf = (__bf16*)ws;                                    // 64 MiB
    __bf16* w2t  = (__bf16*)(ws + (size_t)64 * 1024 * 1024);       // 32 MiB
    float*  xh   = (float*)(ws + (size_t)96 * 1024 * 1024);        // 512 KiB

    // 1) xH = x @ H^T           (one wave per row)
    xh_kernel<<<B_DIM / 8, 256, 0, stream>>>(x, H, xh);
    // 2) W2^T in bf16           (independent of 1)
    w2t_kernel<<<dim3(O_DIM / 32, N_DIM / 32), dim3(32, 8), 0, stream>>>(W2, w2t);
    // 3) h = relu(xH @ G + b1)  -> bf16
    h_kernel<<<(B_DIM * (size_t)N_DIM) / 256, 256, 0, stream>>>(xh, G, b1, hbuf);
    // 4) y = h @ W2 + b2        (async-LDS double-buffered WMMA bf16 GEMM)
    gemm_kernel<<<dim3(O_DIM / 256, B_DIM / 128), 256, 0, stream>>>(hbuf, w2t, b2, y);
}
